// TPUOptimizedSSM_55697135895185
// MI455X (gfx1250) — matrix operand order repaired and checked by reading, not compiled, not run on hardware
//
#include <hip/hip_runtime.h>
#include <hip/hip_bf16.h>

#define B_SZ 8
#define SEQ  1024
#define H    2048
#define M_TOK (B_SZ * SEQ)         // 8192 rows
#define LN_EPS 1e-6f
#define CLIPV 1.0e6f

typedef __attribute__((ext_vector_type(16))) __bf16 bf16x16;
typedef __attribute__((ext_vector_type(8)))  float  f32x8;

__device__ __forceinline__ __bf16 f2bf(float f) { return (__bf16)f; }

// ---------------------------------------------------------------------------
// Weight transpose + f32 -> bf16 convert:  out[n*H + k] = (bf16) in[k*H + n]
// ---------------------------------------------------------------------------
__global__ __launch_bounds__(256)
void transpose_cvt_kernel(const float* __restrict__ in,
                          __bf16* __restrict__ out) {
  __shared__ float tile[32][33];
  const int tx = threadIdx.x;            // 0..31
  const int ty = threadIdx.y;            // 0..7
  const int x0 = blockIdx.x * 32;        // N of input
  const int y0 = blockIdx.y * 32;        // K of input
#pragma unroll
  for (int i = 0; i < 32; i += 8)
    tile[ty + i][tx] = in[(size_t)(y0 + ty + i) * H + (x0 + tx)];
  __syncthreads();
#pragma unroll
  for (int i = 0; i < 32; i += 8)
    out[(size_t)(x0 + ty + i) * H + (y0 + tx)] = f2bf(tile[tx][ty + i]);
}

// ---------------------------------------------------------------------------
// Fused LayerNorm + clip + delta = exp(xn . dt_kernel + dt_bias)
// One 256-thread block per token; 8 elements/thread.
// ---------------------------------------------------------------------------
__device__ __forceinline__ float block_reduce_sum(float v, float* red) {
  const int tid = threadIdx.x;
  red[tid] = v;
  __syncthreads();
#pragma unroll
  for (int off = 128; off > 0; off >>= 1) {
    if (tid < off) red[tid] += red[tid + off];
    __syncthreads();
  }
  const float r = red[0];
  __syncthreads();
  return r;
}

__global__ __launch_bounds__(256)
void ln_delta_kernel(const float* __restrict__ x,
                     const float* __restrict__ ln_scale,
                     const float* __restrict__ ln_bias,
                     const float* __restrict__ dt_kernel,
                     const float* __restrict__ dt_bias,
                     __bf16* __restrict__ xn_bf,
                     float* __restrict__ delta) {
  __shared__ float red[256];
  const int t = blockIdx.x;                       // token index (b*SEQ + s)
  const int tid = threadIdx.x;
  const float* xr = x + (size_t)t * H;

  float v[8];
  float s = 0.f, s2 = 0.f;
#pragma unroll
  for (int j = 0; j < 8; ++j) {
    v[j] = xr[tid + 256 * j];
    s += v[j];
    s2 += v[j] * v[j];
  }
  const float sum  = block_reduce_sum(s,  red);
  const float sum2 = block_reduce_sum(s2, red);
  const float mean = sum * (1.0f / H);
  const float var  = sum2 * (1.0f / H) - mean * mean;
  const float rstd = rsqrtf(var + LN_EPS);

  float dot = 0.f;
#pragma unroll
  for (int j = 0; j < 8; ++j) {
    const int idx = tid + 256 * j;
    float xn = (v[j] - mean) * rstd * ln_scale[idx] + ln_bias[idx];
    xn = fminf(fmaxf(xn, -CLIPV), CLIPV);
    xn_bf[(size_t)t * H + idx] = f2bf(xn);
    dot += xn * dt_kernel[idx];
  }
  const float d = block_reduce_sum(dot, red);
  if (tid == 0) delta[t] = __expf(d + dt_bias[0]);
}

// ---------------------------------------------------------------------------
// Diagonal-decay scan:  state = state*exp(delta_t * A_h) + U_t   (serial in S)
// One thread per (b,h) channel.  Writes bf16 states + final_state (f32).
// ---------------------------------------------------------------------------
__global__ __launch_bounds__(256)
void scan_kernel(const float* __restrict__ U,
                 const float* __restrict__ delta,
                 const float* __restrict__ Avec,
                 const float* __restrict__ context,
                 __bf16* __restrict__ Sbf,
                 float* __restrict__ final_state) {
  const int tid = blockIdx.x * blockDim.x + threadIdx.x;   // 0 .. B*H-1
  const int b = tid / H;
  const int h = tid - b * H;
  const float a = Avec[h];
  float st = context[(size_t)b * H + h];
  const float* up = U + (size_t)b * SEQ * H + h;
  const float* dp = delta + (size_t)b * SEQ;
  __bf16* sp = Sbf + (size_t)b * SEQ * H + h;
  for (int s = 0; s < SEQ; ++s) {
    const float dec = __expf(dp[s] * a);
    st = st * dec + up[(size_t)s * H];
    sp[(size_t)s * H] = f2bf(st);
  }
  final_state[tid] = st;
}

// ---------------------------------------------------------------------------
// WMMA GEMM:  C[M x N] = A[M x K](bf16, row-major) * W[N x K](bf16, K-contig)
// Wave tile 64x64 (4x4 of v_wmma_f32_16x16x32_bf16), 8 waves -> 128x256 block.
// K, N are compile-time (2048) so all fragment loads use immediate offsets
// from just two base pointers.
// MODE 0: store f32        MODE 1: store f32 + bf16
// MODE 2: g = acc + bias[n];  out = y32 * sigmoid(g)     (fused GLU epilogue)
// ---------------------------------------------------------------------------
template <int MODE>
__global__ __launch_bounds__(256, 1)
void gemm_wmma_kernel(const __bf16* __restrict__ Abf,
                      const __bf16* __restrict__ Wbf,
                      float* __restrict__ out32,
                      __bf16* __restrict__ outbf,
                      const float* __restrict__ y32,
                      const float* __restrict__ bias) {
  constexpr int K = H;                          // 2048
  constexpr int N = H;                          // 2048
  const int lane   = threadIdx.x & 31;
  const int waveId = threadIdx.x >> 5;          // 0..7
  const int wm = waveId >> 2;                   // 0..1
  const int wn = waveId & 3;                    // 0..3
  const int mBase = blockIdx.y * 128 + wm * 64;
  const int nBase = blockIdx.x * 256 + wn * 64;

  const int lm = lane & 15;                     // row/col within 16
  const int lh = lane >> 4;                     // half select

  // Single base pointer per matrix; 16-row blocks addressed via immediates.
  const __bf16* aptr = Abf + (size_t)(mBase + lm) * K + lh * 8;
  const __bf16* bptr = Wbf + (size_t)(nBase + lm) * K + lh * 8;

  f32x8 acc[4][4];
#pragma unroll
  for (int i = 0; i < 4; ++i)
#pragma unroll
    for (int j = 0; j < 4; ++j) acc[i][j] = (f32x8)0.f;

  for (int k0 = 0; k0 < K; k0 += 32) {
    bf16x16 a[4], b[4];
#pragma unroll
    for (int i = 0; i < 4; ++i) {
      union { uint4 u[2]; bf16x16 v; } t;
      t.u[0] = *(const uint4*)(aptr + i * (16 * K));
      t.u[1] = *(const uint4*)(aptr + i * (16 * K) + 16);
      a[i] = t.v;
    }
#pragma unroll
    for (int j = 0; j < 4; ++j) {
      union { uint4 u[2]; bf16x16 v; } t;
      t.u[0] = *(const uint4*)(bptr + j * (16 * K));
      t.u[1] = *(const uint4*)(bptr + j * (16 * K) + 16);
      b[j] = t.v;
    }
#pragma unroll
    for (int i = 0; i < 4; ++i)
#pragma unroll
      for (int j = 0; j < 4; ++j)
        acc[i][j] = __builtin_amdgcn_wmma_f32_16x16x32_bf16(
            false, a[i], false, b[j], (short)0, acc[i][j], false, false);
    aptr += 32;
    bptr += 32;
  }

  // C/D layout: lane l, VGPR r -> row = 8*(l>>4) + r, col = l & 15
#pragma unroll
  for (int i = 0; i < 4; ++i) {
#pragma unroll
    for (int j = 0; j < 4; ++j) {
      const int col = nBase + 16 * j + lm;
      const size_t base = (size_t)(mBase + 16 * i + 8 * lh) * N + col;
#pragma unroll
      for (int r = 0; r < 8; ++r) {
        const size_t idx = base + (size_t)r * N;
        const float v = acc[i][j][r];
        if (MODE == 0) {
          out32[idx] = v;
        } else if (MODE == 1) {
          out32[idx] = v;
          outbf[idx] = f2bf(v);
        } else {
          const float g = v + bias[col];
          const float sig = 1.0f / (1.0f + __expf(-g));
          out32[idx] = y32[idx] * sig;
        }
      }
    }
  }
}

// ---------------------------------------------------------------------------
// Launch
// ---------------------------------------------------------------------------
extern "C" void kernel_launch(void* const* d_in, const int* in_sizes, int n_in,
                              void* d_out, int out_size, void* d_ws, size_t ws_size,
                              hipStream_t stream) {
  const float* x        = (const float*)d_in[0];
  const float* context  = (const float*)d_in[1];
  const float* Avec     = (const float*)d_in[2];
  const float* Bm       = (const float*)d_in[3];
  const float* Cm       = (const float*)d_in[4];
  const float* dtk      = (const float*)d_in[5];
  const float* dtb      = (const float*)d_in[6];
  const float* gk       = (const float*)d_in[7];
  const float* gb       = (const float*)d_in[8];
  const float* lns      = (const float*)d_in[9];
  const float* lnb      = (const float*)d_in[10];
  float* out            = (float*)d_out;

  char* ws = (char*)d_ws;
  const size_t SZ_XN    = (size_t)M_TOK * H * 2;   // 32 MB bf16
  const size_t SZ_DELTA = (size_t)M_TOK * 4;       // 32 KB
  const size_t SZ_W     = (size_t)H * H * 2;       // 8 MB bf16
  const size_t SZ_U     = (size_t)M_TOK * H * 4;   // 64 MB f32

  size_t off = 0;
  __bf16* xn_bf = (__bf16*)(ws + off); off += SZ_XN;
  float*  delta = (float*)(ws + off);  off += SZ_DELTA;
  __bf16* Bt    = (__bf16*)(ws + off); off += SZ_W;
  __bf16* Ct    = (__bf16*)(ws + off); off += SZ_W;
  __bf16* Gt    = (__bf16*)(ws + off); off += SZ_W;
  float*  U     = (float*)(ws + off);  off += SZ_U;
  __bf16* Sbf   = (__bf16*)(ws + off); off += SZ_XN;
  // aliases (producers/consumers do not overlap in time):
  float*  Yf32  = U;        // reuse U after scan consumed it
  __bf16* Ybf   = xn_bf;    // reuse xn after GEMM1 consumed it
  float*  final_state = out + (size_t)M_TOK * H;

  // 1) weights -> bf16, transposed to [N][K]
  {
    dim3 blk(32, 8), grd(H / 32, H / 32);
    transpose_cvt_kernel<<<grd, blk, 0, stream>>>(Bm, Bt);
    transpose_cvt_kernel<<<grd, blk, 0, stream>>>(Cm, Ct);
    transpose_cvt_kernel<<<grd, blk, 0, stream>>>(gk, Gt);
  }
  // 2) LayerNorm + clip + delta
  ln_delta_kernel<<<M_TOK, 256, 0, stream>>>(x, lns, lnb, dtk, dtb, xn_bf, delta);
  // 3) U = xn @ Bm
  {
    dim3 blk(256), grd(H / 256, M_TOK / 128);
    gemm_wmma_kernel<0><<<grd, blk, 0, stream>>>(xn_bf, Bt, U, nullptr, nullptr,
                                                 nullptr);
  }
  // 4) diagonal scan
  scan_kernel<<<(B_SZ * H) / 256, 256, 0, stream>>>(U, delta, Avec, context, Sbf,
                                                    final_state);
  // 5) Y = states @ Cm   (f32 + bf16 copies)
  {
    dim3 blk(256), grd(H / 256, M_TOK / 128);
    gemm_wmma_kernel<1><<<grd, blk, 0, stream>>>(Sbf, Ct, Yf32, Ybf, nullptr,
                                                 nullptr);
  }
  // 6) out = Y * sigmoid(Y @ gate_kernel + gate_bias)
  {
    dim3 blk(256), grd(H / 256, M_TOK / 128);
    gemm_wmma_kernel<2><<<grd, blk, 0, stream>>>(Ybf, Gt, out, nullptr, Yf32,
                                                 gb);
  }
}
